// GraphRegressorV2_77378130805145
// MI455X (gfx1250) — matrix operand order repaired and checked by reading, not compiled, not run on hardware
//
#include <hip/hip_runtime.h>
#include <hip/hip_bf16.h>

typedef __attribute__((ext_vector_type(16))) __bf16 v16bf;
typedef __attribute__((ext_vector_type(8)))  float  v8f;

#define HDIM 256

// ---------------- WMMA fragment helpers (ISA 7.12.2 layouts, wave32) ---------

// A fragment: 16x32 bf16 tile starting at column kbase of a row-major LDS tile.
// lane m = lane&15 ; lanes 0-15 hold K {0..7,16..23}, lanes 16-31 hold {8..15,24..31}
// Per lane this is two contiguous 16B chunks -> compiler emits 2x ds_load_b128.
__device__ __forceinline__ v16bf load_a_frag(const __bf16* At, int lda, int lane, int kbase) {
  int m  = lane & 15;
  int hi = (lane >> 4) & 1;
  v16bf a;
#pragma unroll
  for (int e = 0; e < 16; ++e) {
    int d = e >> 1, h = e & 1;
    int k = (d < 4) ? ((hi ? 8 : 0) + 2 * d + h)
                    : ((hi ? 24 : 16) + 2 * (d - 4) + h);
    a[e] = At[m * lda + kbase + k];
  }
  return a;
}

// B fragment from pre-swizzled weights: one contiguous 32B chunk per lane
// (coalesced 1KB per wave -> 2x global_load_b128).
__device__ __forceinline__ v16bf load_b_swz(const __bf16* Wswz, int KB, int ct, int kb,
                                            int lane) {
  return *(const v16bf*)(Wswz + (((size_t)(ct * KB + kb) * 32 + lane) << 4));
}

// ---------------- utility kernels -------------------------------------------

__global__ void zero_f(float* p, int n) {
  int i = blockIdx.x * blockDim.x + threadIdx.x;
  if (i < n) p[i] = 0.f;
}

__global__ void fill_f(float* p, float v, int n) {
  int i = blockIdx.x * blockDim.x + threadIdx.x;
  if (i < n) p[i] = v;
}

// Convert f32 row-major W[K][Ncols] into bf16 fragment-major swizzled layout:
// dst[(((ct*KB + kb)*32) + lane)*16 + e] = W[kb*32 + (lane>=16)*8 + (e&7) + (e>>3)*16][ct*16 + (lane&15)]
__global__ void swizzleW(const float* __restrict__ W, __bf16* __restrict__ dst,
                         int K, int Ncols) {
  int i = blockIdx.x * blockDim.x + threadIdx.x;
  if (i >= K * Ncols) return;
  int e    = i & 15;
  int lane = (i >> 4) & 31;
  int blk  = i >> 9;            // ct*KB + kb
  int KB   = K >> 5;
  int ct   = blk / KB, kb = blk % KB;
  int n = ct * 16 + (lane & 15);
  int k = kb * 32 + ((lane >> 4) & 1) * 8 + (e & 7) + (e >> 3) * 16;
  dst[i] = (__bf16)W[k * Ncols + n];
}

// ---------------- node embedding: relu(x[:, :3]@Wc+bc) || relu(x[:,3:]@Wn+bn)
__global__ void node_embed(const float* __restrict__ x,
                           const float* __restrict__ coord_W, const float* __restrict__ coord_b,
                           const float* __restrict__ node_W,  const float* __restrict__ node_b,
                           float* __restrict__ x_emb, int Nn) {
  int i = blockIdx.x * blockDim.x + threadIdx.x;
  if (i >= Nn * HDIM) return;
  int nd = i >> 8, c = i & 255;
  const float* xr = x + nd * 9;
  float acc;
  if (c < 128) {
    acc = coord_b[c];
#pragma unroll
    for (int j = 0; j < 3; ++j) acc += xr[j] * coord_W[j * 128 + c];
  } else {
    int cc = c - 128;
    acc = node_b[cc];
#pragma unroll
    for (int j = 0; j < 6; ++j) acc += xr[3 + j] * node_W[j * 128 + cc];
  }
  x_emb[i] = fmaxf(acc, 0.f);
}

// ---------------- degree / norm ---------------------------------------------
__global__ void deg_kernel(const int* __restrict__ ei, float* deg, int E_) {
  int e = blockIdx.x * blockDim.x + threadIdx.x;
  if (e < E_) atomicAdd(&deg[ei[E_ + e]], 1.f);  // col = edge_index[1]
}

__global__ void dnorm_kernel(const float* __restrict__ deg, float* dn, int n) {
  int i = blockIdx.x * blockDim.x + threadIdx.x;
  if (i < n) dn[i] = rsqrtf(deg[i]);
}

// ---------------- edge embedding + projection + scatter (WMMA) --------------
// 16 edges per block. Stage 1: relu(edge_attr@edge_W+b) -> LDS bf16 [16x128].
// Stage 2: [16x128]@[128x256] via v_wmma_f32_16x16x32_bf16 with register-cached
// A fragments and swizzled B. Branchless atomic scatter: padded tail edges are
// redirected to dummy row Nn of node_msg (allocated with spare rows, never read).
__global__ void edge_kernel(const float* __restrict__ edge_attr,
                            const int* __restrict__ ei,
                            const float* __restrict__ edge_W, const float* __restrict__ edge_b,
                            const __bf16* __restrict__ projWswz, const float* __restrict__ proj_b,
                            float* __restrict__ node_msg, int E_, int Nn) {
  __shared__ float  sW[8 * 128];
  __shared__ float  sEA[16 * 8];
  __shared__ __bf16 At[16][128 + 8];
  __shared__ alignas(16) int sDst[16];
  int eb  = blockIdx.x * 16;
  int tid = threadIdx.x;

  for (int i = tid; i < 8 * 128; i += 256) sW[i] = edge_W[i];
  if (tid < 16 * 8) {
    int e = eb + (tid >> 3);
    sEA[tid] = (e < E_) ? edge_attr[e * 8 + (tid & 7)] : 0.f;
  }
  if (tid < 16) {
    int e = eb + tid;
    sDst[tid] = (e < E_) ? ei[E_ + e] : Nn;   // dummy row for padding
  }
  // prefetch next block's edge_attr while this block computes
  if (tid < 128 && (eb + 16) * 8 + tid < E_ * 8)
    __builtin_prefetch(edge_attr + (eb + 16) * 8 + tid, 0, 0);
  __syncthreads();

  for (int i = tid; i < 16 * 128; i += 256) {
    int r = i >> 7, c = i & 127;
    float acc = edge_b[c];
#pragma unroll
    for (int j = 0; j < 8; ++j) acc += sEA[r * 8 + j] * sW[j * 128 + c];
    At[r][c] = (__bf16)fmaxf(acc, 0.f);
  }
  __syncthreads();

  int wave = tid >> 5, lane = tid & 31;
  int mo = ((lane >> 4) & 1) * 8;
  // 8 destination rows for this lane-half: two ds_load_b128
  int dsts[8];
  *(int4*)&dsts[0] = *(const int4*)&sDst[mo];
  *(int4*)&dsts[4] = *(const int4*)&sDst[mo + 4];

  v16bf afrag[4];
#pragma unroll
  for (int kb = 0; kb < 4; ++kb) afrag[kb] = load_a_frag(&At[0][0], 128 + 8, lane, kb * 32);

#pragma unroll
  for (int t = 0; t < 2; ++t) {
    int ct = wave * 2 + t;
    v8f c = {};
#pragma unroll
    for (int kb = 0; kb < 4; ++kb) {
      v16bf b = load_b_swz(projWswz, 4, ct, kb, lane);
      c = __builtin_amdgcn_wmma_f32_16x16x32_bf16(false, afrag[kb], false, b, (short)0, c,
                                                  false, false);
    }
    int n  = ct * 16 + (lane & 15);
    float pb = proj_b[n];
#pragma unroll
    for (int r = 0; r < 8; ++r)
      atomicAdd(&node_msg[dsts[r] * HDIM + n], c[r] + pb);
  }
}

// ---------------- GCN GEMM: h = (x_emb + node_msg) @ W  (WMMA) --------------
__global__ void gcn_gemm(const float* __restrict__ x_emb, const float* __restrict__ node_msg,
                         const __bf16* __restrict__ Wswz, float* __restrict__ h, int Nn) {
  __shared__ __bf16 At[16][HDIM + 8];
  int rb  = blockIdx.x * 16;
  int tid = threadIdx.x;
  for (int i = tid; i < 16 * HDIM; i += 256) {
    int r = i >> 8, c = i & 255;
    int row = rb + r;
    At[r][c] = (row < Nn) ? (__bf16)(x_emb[row * HDIM + c] + node_msg[row * HDIM + c])
                          : (__bf16)0.f;
  }
  // prefetch next row-block's inputs (16 rows * 1KB each = 256 threads * 64B)
  if (rb + 16 < Nn) {
    __builtin_prefetch(x_emb    + (size_t)(rb + 16) * HDIM + tid * 16, 0, 0);
    __builtin_prefetch(node_msg + (size_t)(rb + 16) * HDIM + tid * 16, 0, 0);
  }
  __syncthreads();

  int wave = tid >> 5, lane = tid & 31;
  v16bf afrag[8];
#pragma unroll
  for (int kb = 0; kb < 8; ++kb) afrag[kb] = load_a_frag(&At[0][0], HDIM + 8, lane, kb * 32);

#pragma unroll
  for (int t = 0; t < 2; ++t) {
    int ct = wave * 2 + t;
    v8f c = {};
#pragma unroll
    for (int kb = 0; kb < 8; ++kb) {
      v16bf b = load_b_swz(Wswz, 8, ct, kb, lane);
      c = __builtin_amdgcn_wmma_f32_16x16x32_bf16(false, afrag[kb], false, b, (short)0, c,
                                                  false, false);
    }
    int n  = ct * 16 + (lane & 15);
    int mo = ((lane >> 4) & 1) * 8;
#pragma unroll
    for (int r = 0; r < 8; ++r) {
      int row = rb + r + mo;
      if (row < Nn) h[row * HDIM + n] = c[r];
    }
  }
}

// ---------------- GCN propagation: accum[col] += h[row]*dn[row]*dn[col] -----
__global__ void gcn_scatter(const float* __restrict__ h, const int* __restrict__ ei,
                            const float* __restrict__ dn, float* __restrict__ accum, int E_) {
  int i = blockIdx.x * blockDim.x + threadIdx.x;   // E*64 threads, 4 floats each
  if (i >= E_ * 64) return;
  int e = i >> 6, q = (i & 63) * 4;
  int r = ei[e], c = ei[E_ + e];
  float s = dn[r] * dn[c];
  const float4 hv = *(const float4*)(h + r * HDIM + q);
  float* a = accum + c * HDIM + q;
  atomicAdd(a + 0, hv.x * s);
  atomicAdd(a + 1, hv.y * s);
  atomicAdd(a + 2, hv.z * s);
  atomicAdd(a + 3, hv.w * s);
}

// self-loop term + bias + (optional) relu -> writes x_emb in place
__global__ void gcn_finalize(const float* __restrict__ accum, const float* __restrict__ h,
                             const float* __restrict__ dn, const float* __restrict__ bias,
                             float* __restrict__ x_emb, int Nn, int do_relu) {
  int i = blockIdx.x * blockDim.x + threadIdx.x;
  if (i >= Nn * HDIM) return;
  int nd = i >> 8, c = i & 255;
  float d = dn[nd];
  float v = accum[i] + h[i] * d * d + bias[c];
  x_emb[i] = do_relu ? fmaxf(v, 0.f) : v;
}

// ---------------- attention scores: wave per node ---------------------------
__global__ void score_kernel(const float* __restrict__ x_emb, const float* __restrict__ aw,
                             const float* __restrict__ ab, float* __restrict__ scores, int Nn) {
  int wave = threadIdx.x >> 5, lane = threadIdx.x & 31;
  int nd = blockIdx.x * 8 + wave;
  if (nd >= Nn) return;
  float s = 0.f;
#pragma unroll
  for (int j = 0; j < 8; ++j) {
    int c = lane + 32 * j;
    s += x_emb[nd * HDIM + c] * aw[c];
  }
#pragma unroll
  for (int off = 16; off > 0; off >>= 1) s += __shfl_down(s, off, 32);
  if (lane == 0) scores[nd] = s + ab[0];
}

// ---------------- attention pooling: block per graph ------------------------
__global__ void attn_pool(const float* __restrict__ scores, const int* __restrict__ batch,
                          const float* __restrict__ x_emb, float* __restrict__ graph_emb,
                          int Nn) {
  int g = blockIdx.x, tid = threadIdx.x;
  __shared__ float red[256];
  __shared__ float acc[256];
  float m = -3.0e38f;
  for (int n = tid; n < Nn; n += 256)
    if (batch[n] == g) m = fmaxf(m, scores[n]);
  red[tid] = m; __syncthreads();
  for (int s = 128; s > 0; s >>= 1) {
    if (tid < s) red[tid] = fmaxf(red[tid], red[tid + s]);
    __syncthreads();
  }
  m = red[0]; __syncthreads();

  float sum = 0.f;
  for (int n = tid; n < Nn; n += 256)
    if (batch[n] == g) sum += __expf(scores[n] - m);
  red[tid] = sum; __syncthreads();
  for (int s = 128; s > 0; s >>= 1) {
    if (tid < s) red[tid] += red[tid + s];
    __syncthreads();
  }
  float denom = red[0];
  acc[tid] = 0.f; __syncthreads();

  for (int n = tid; n < Nn; n += 256) {
    if (batch[n] == g) {
      float w = __expf(scores[n] - m) / denom;
      const float* xr = x_emb + n * HDIM;
      for (int c = 0; c < HDIM; ++c) atomicAdd(&acc[c], w * xr[c]);
    }
  }
  __syncthreads();
  graph_emb[g * HDIM + tid] = acc[tid];
}

// ---------------- head: global MLP + concat + fc_out (block per graph) ------
__global__ void head_kernel(const float* __restrict__ gx,
                            const float* __restrict__ W1, const float* __restrict__ b1,
                            const float* __restrict__ W2, const float* __restrict__ b2,
                            const float* __restrict__ graph_emb,
                            const float* __restrict__ fW1, const float* __restrict__ fb1,
                            const float* __restrict__ fW2, const float* __restrict__ fb2,
                            float* __restrict__ out) {
  int g = blockIdx.x, tid = threadIdx.x;
  __shared__ float gh[256];
  __shared__ float cat[512];
  __shared__ float red[256];
  float a = b1[tid];
#pragma unroll
  for (int j = 0; j < 16; ++j) a += gx[g * 16 + j] * W1[j * 256 + tid];
  gh[tid] = fmaxf(a, 0.f);
  __syncthreads();
  float b = b2[tid];
  for (int j = 0; j < 256; ++j) b += gh[j] * W2[j * 256 + tid];
  cat[tid]       = graph_emb[g * 256 + tid];
  cat[256 + tid] = b;
  __syncthreads();
  float c = fb1[tid];
  for (int j = 0; j < 512; ++j) c += cat[j] * fW1[j * 256 + tid];
  red[tid] = fmaxf(c, 0.f) * fW2[tid];
  __syncthreads();
  for (int s = 128; s > 0; s >>= 1) {
    if (tid < s) red[tid] += red[tid + s];
    __syncthreads();
  }
  if (tid == 0) out[g] = red[0] + fb2[0];
}

// ---------------- host launch ----------------------------------------------

extern "C" void kernel_launch(void* const* d_in, const int* in_sizes, int n_in,
                              void* d_out, int out_size, void* d_ws, size_t ws_size,
                              hipStream_t stream) {
  const float* x        = (const float*)d_in[0];
  const int*   ei       = (const int*)  d_in[1];
  const int*   batch    = (const int*)  d_in[2];
  const float* edge_at  = (const float*)d_in[3];
  const float* gx       = (const float*)d_in[4];
  const float* coord_W  = (const float*)d_in[5];
  const float* coord_b  = (const float*)d_in[6];
  const float* node_W   = (const float*)d_in[7];
  const float* node_b   = (const float*)d_in[8];
  const float* edge_W   = (const float*)d_in[9];
  const float* edge_b   = (const float*)d_in[10];
  const float* proj_W   = (const float*)d_in[11];
  const float* proj_b   = (const float*)d_in[12];
  const float* gcn_W    = (const float*)d_in[13];
  const float* gcn_b    = (const float*)d_in[14];
  const float* attn_w   = (const float*)d_in[15];
  const float* attn_b   = (const float*)d_in[16];
  const float* gmlp_W1  = (const float*)d_in[17];
  const float* gmlp_b1  = (const float*)d_in[18];
  const float* gmlp_W2  = (const float*)d_in[19];
  const float* gmlp_b2  = (const float*)d_in[20];
  const float* fc_W1    = (const float*)d_in[21];
  const float* fc_b1    = (const float*)d_in[22];
  const float* fc_W2    = (const float*)d_in[23];
  const float* fc_b2    = (const float*)d_in[24];
  float* out = (float*)d_out;

  const int N = in_sizes[2];          // batch has N entries
  const int E = in_sizes[1] / 2;      // edge_index is (2, E)
  const int G = in_sizes[4] / 16;     // global_x is (G, 16)
  const int L = 3;

  // workspace layout (256B aligned)
  char* ws = (char*)d_ws;
  size_t off = 0;
  auto take = [&](size_t bytes) -> char* {
    char* p = ws + off;
    off = (off + bytes + 255) & ~(size_t)255;
    return p;
  };
  float*  x_emb     = (float*)take((size_t)N * HDIM * 4);
  float*  node_msg  = (float*)take((size_t)(N + 16) * HDIM * 4);  // +16 dummy rows
  float*  hbuf      = (float*)take((size_t)N * HDIM * 4);
  float*  accum     = (float*)take((size_t)N * HDIM * 4);
  float*  deg       = (float*)take((size_t)N * 4);
  float*  dnorm     = (float*)take((size_t)N * 4);
  float*  scores    = (float*)take((size_t)N * 4);
  float*  graph_emb = (float*)take((size_t)G * HDIM * 4);
  __bf16* projWswz  = (__bf16*)take((size_t)128 * HDIM * 2);
  __bf16* gcnWswz   = (__bf16*)take((size_t)L * HDIM * HDIM * 2);
  (void)n_in; (void)out_size; (void)ws_size;

  const int T = 256;
  auto blocks = [](long long n, int t) { return (unsigned)((n + t - 1) / t); };

  // init + weight swizzle (bf16 fragment-major for coalesced B loads)
  fill_f<<<blocks(N, T), T, 0, stream>>>(deg, 1.0f, N);                 // self-loops
  zero_f<<<blocks((long long)N * HDIM, T), T, 0, stream>>>(node_msg, N * HDIM);
  swizzleW<<<blocks(128 * HDIM, T), T, 0, stream>>>(proj_W, projWswz, 128, HDIM);
  for (int i = 0; i < L; ++i)
    swizzleW<<<blocks(HDIM * HDIM, T), T, 0, stream>>>(
        gcn_W + (size_t)i * HDIM * HDIM, gcnWswz + (size_t)i * HDIM * HDIM, HDIM, HDIM);

  node_embed<<<blocks((long long)N * HDIM, T), T, 0, stream>>>(
      x, coord_W, coord_b, node_W, node_b, x_emb, N);

  deg_kernel<<<blocks(E, T), T, 0, stream>>>(ei, deg, E);
  dnorm_kernel<<<blocks(N, T), T, 0, stream>>>(deg, dnorm, N);

  edge_kernel<<<blocks((long long)E, 16), T, 0, stream>>>(
      edge_at, ei, edge_W, edge_b, projWswz, proj_b, node_msg, E, N);

  for (int i = 0; i < L; ++i) {
    zero_f<<<blocks((long long)N * HDIM, T), T, 0, stream>>>(accum, N * HDIM);
    gcn_gemm<<<blocks((long long)N, 16), T, 0, stream>>>(
        x_emb, node_msg, gcnWswz + (size_t)i * HDIM * HDIM, hbuf, N);
    gcn_scatter<<<blocks((long long)E * 64, T), T, 0, stream>>>(hbuf, ei, dnorm, accum, E);
    gcn_finalize<<<blocks((long long)N * HDIM, T), T, 0, stream>>>(
        accum, hbuf, dnorm, gcn_b + i * HDIM, x_emb, N, (i < L - 1) ? 1 : 0);
  }

  score_kernel<<<blocks((long long)N, 8), T, 0, stream>>>(x_emb, attn_w, attn_b, scores, N);
  attn_pool<<<G, T, 0, stream>>>(scores, batch, x_emb, graph_emb, N);
  head_kernel<<<G, T, 0, stream>>>(gx, gmlp_W1, gmlp_b1, gmlp_W2, gmlp_b2,
                                   graph_emb, fc_W1, fc_b1, fc_W2, fc_b2, out);
}
